// TriangleAttention_34187939676339
// MI455X (gfx1250) — matrix operand order repaired and checked by reading, not compiled, not run on hardware
//
#include <hip/hip_runtime.h>
#include <hip/hip_bf16.h>

// ---------------------------------------------------------------------------
// Triangle attention (outgoing), L=320, C=128, H=4, D=32, fp32 in/out.
// All matmuls on v_wmma_f32_16x16x32_bf16 (bf16 inputs, f32 accumulate),
// f32 softmax with v_rcp_f32 normalization. 4 kernels:
//   k_weights: fp32->bf16 transposed weights
//   k_proj:    fused Q/K/V/Gate projection (16-row tiles, 8 waves/block)
//   k_attn:    per-(i,h) attention, 2 waves/block, LDS-staged scores
//   k_out:     attn_out @ Wo + bo, * gate
// ---------------------------------------------------------------------------

#define L_DIM 320
#define C_DIM 128
#define H_DIM 4
#define D_DIM 32
#define NROWS (L_DIM * L_DIM)          // 102400
#define NTILES (NROWS / 16)            // 6400

typedef __attribute__((ext_vector_type(16))) __bf16 v16bf;
typedef __attribute__((ext_vector_type(8)))  float  v8f;

union FragB {
    v16bf v;
    unsigned int u[8];
};

// A-matrix (16x32 bf16) per-lane K offsets: lanes 0-15 hold K 0..15,
// lanes 16-31 hold K 16..31; VGPR j holds a consecutive K pair.
__device__ __forceinline__ constexpr int kmapA(int j, int half) {
    return 16 * half + 2 * j;
}
// B-matrix (32x16 bf16): lanes 0-15 hold K 0..15, lanes 16-31 hold K 16..31.
__device__ __forceinline__ constexpr int kmapB(int j, int half) {
    return 16 * half + 2 * j;
}

__device__ __forceinline__ v8f wmma_bf16(const FragB& a, const FragB& b, v8f c) {
    return __builtin_amdgcn_wmma_f32_16x16x32_bf16(
        /*neg_a=*/false, a.v, /*neg_b=*/false, b.v,
        /*c_mod=*/(short)0, c, /*reuse_a=*/false, /*reuse_b=*/false);
}

// ---------------------------------------------------------------------------
// Kernel 0: weights fp32 -> bf16, TRANSPOSED to [c_out][c_in] so B-fragment
// K-pairs are contiguous dwords. Slots: 0=Wq 1=Wk 2=Wv 3=Wg 4=Wo.
// ---------------------------------------------------------------------------
__global__ void k_weights(const float* __restrict__ Wq, const float* __restrict__ Wk,
                          const float* __restrict__ Wv, const float* __restrict__ Wg,
                          const float* __restrict__ Wo, __bf16* __restrict__ Wp) {
    int n  = blockIdx.x;        // c_out
    int m  = blockIdx.y;        // weight slot
    int kk = threadIdx.x;       // c_in
    const float* src = (m == 0) ? Wq : (m == 1) ? Wk : (m == 2) ? Wv : (m == 3) ? Wg : Wo;
    Wp[(size_t)m * 16384 + (size_t)n * 128 + kk] = (__bf16)src[(size_t)kk * 128 + n];
}

// ---------------------------------------------------------------------------
// Kernel 1: fused Q/K/V/Gate projection.
// One block = one 16-row tile of pair_repr (bf16 in LDS, b128 load path).
// 8 waves: wave w handles weight m=w/2, N-half (w&1)*64. A-fragments hoisted
// (loaded once per wave, reused for all 4 N-tiles). 16 WMMA / wave.
// Q pre-scaled by 1/sqrt(D); V stored transposed [i][h][d][j]; gate=sigmoid.
// ---------------------------------------------------------------------------
__global__ void __launch_bounds__(256) k_proj(
    const float* __restrict__ X, const __bf16* __restrict__ Wp,
    const float* __restrict__ bq, const float* __restrict__ bk,
    const float* __restrict__ bv, const float* __restrict__ bg,
    __bf16* __restrict__ qb, __bf16* __restrict__ kb,
    __bf16* __restrict__ vT, float* __restrict__ gate) {

    __shared__ __bf16 Xs[16 * 128];

    const int rowbase = blockIdx.x * 16;
    {   // 8 contiguous fp32 per thread -> global b128 loads + ds b128 store
        const int e0 = threadIdx.x * 8;
        const float4* xs = (const float4*)(X + (size_t)rowbase * 128);
        float4 a = xs[threadIdx.x * 2];
        float4 b = xs[threadIdx.x * 2 + 1];
        Xs[e0 + 0] = (__bf16)a.x; Xs[e0 + 1] = (__bf16)a.y;
        Xs[e0 + 2] = (__bf16)a.z; Xs[e0 + 3] = (__bf16)a.w;
        Xs[e0 + 4] = (__bf16)b.x; Xs[e0 + 5] = (__bf16)b.y;
        Xs[e0 + 6] = (__bf16)b.z; Xs[e0 + 7] = (__bf16)b.w;
    }
    __syncthreads();

    const int wave = threadIdx.x >> 5;
    const int lane = threadIdx.x & 31;
    const int lane15 = lane & 15;
    const int half = lane >> 4;
    const int m = wave >> 1;                 // 0=q 1=k 2=v 3=g
    const int nh = (wave & 1) * 64;
    const __bf16* W = Wp + (size_t)m * 16384;
    const float* bias = (m == 0) ? bq : (m == 1) ? bk : (m == 2) ? bv : bg;
    const int i = rowbase / L_DIM;
    const int jbase = rowbase % L_DIM;

    // Hoisted A-fragments: one per K-chunk, shared by all 4 N-tiles.
    FragB af[4];
#pragma unroll
    for (int kt = 0; kt < 4; ++kt)
#pragma unroll
        for (int j = 0; j < 8; ++j)
            af[kt].u[j] = *(const unsigned int*)&Xs[lane15 * 128 + kt * 32 + kmapA(j, half)];

    for (int nt = 0; nt < 4; ++nt) {
        const int c = nh + nt * 16 + lane15;       // output channel
        const float b = bias[c];
        v8f acc = {};
#pragma unroll
        for (int z = 0; z < 8; ++z) acc[z] = b;

#pragma unroll
        for (int kt = 0; kt < 4; ++kt) {
            FragB bf;
#pragma unroll
            for (int j = 0; j < 8; ++j)
                bf.u[j] = *(const unsigned int*)&W[(size_t)c * 128 + kt * 32 + kmapB(j, half)];
            acc = wmma_bf16(af[kt], bf, acc);
        }

        if (m == 0) {             // Q, pre-scaled by 1/sqrt(32)
#pragma unroll
            for (int vv = 0; vv < 8; ++vv)
                qb[(size_t)(rowbase + vv + 8 * half) * 128 + c] =
                    (__bf16)(acc[vv] * 0.17677669529663687f);
        } else if (m == 1) {      // K
#pragma unroll
            for (int vv = 0; vv < 8; ++vv)
                kb[(size_t)(rowbase + vv + 8 * half) * 128 + c] = (__bf16)acc[vv];
        } else if (m == 2) {      // V transposed: [i][h][d][j]
            const int hh = c >> 5, dd = c & 31;
#pragma unroll
            for (int vv = 0; vv < 8; ++vv)
                vT[(size_t)((i * H_DIM + hh) * D_DIM + dd) * L_DIM + jbase + vv + 8 * half] =
                    (__bf16)acc[vv];
        } else {                  // gate = sigmoid(z) via v_exp + v_rcp
#pragma unroll
            for (int vv = 0; vv < 8; ++vv)
                gate[(size_t)(rowbase + vv + 8 * half) * 128 + c] =
                    __builtin_amdgcn_rcpf(1.0f + __expf(-acc[vv]));
        }
    }
}

// ---------------------------------------------------------------------------
// Kernel 2: attention per (i, h). Block = 64 threads = 2 waves; each wave owns
// a 16-row q tile. Scores: 20 WMMA (K=D=32 -> one WMMA per 16-col j tile) into
// padded LDS (stride 324 avoids 64-bank conflicts). f32 softmax (v_rcp_f32),
// exp stored as bf16 P tile, 1/sum folded into output. P@V: kt-outer loop,
// P fragment loaded once, feeding both N-tile accumulators (halves ds loads).
// ---------------------------------------------------------------------------
#define SP 324   // padded score/P row stride (324 % 64 == 4)

__global__ void __launch_bounds__(64) k_attn(
    const __bf16* __restrict__ qb, const __bf16* __restrict__ kb,
    const __bf16* __restrict__ vT, __bf16* __restrict__ ob) {

    __shared__ float  S[2][16][SP];
    __shared__ __bf16 P[2][16][SP];
    __shared__ float  Inv[2][16];

    const int wave = threadIdx.x >> 5;
    const int lane = threadIdx.x & 31;
    const int lane15 = lane & 15;
    const int half = lane >> 4;
    const int i = blockIdx.z;
    const int h = blockIdx.y;
    const int qbase = blockIdx.x * 32 + wave * 16;

    // Q fragment for this wave's 16 rows (loaded once).
    FragB qf;
    const __bf16* qrow = qb + (size_t)(i * L_DIM + qbase + lane15) * C_DIM + h * D_DIM;
#pragma unroll
    for (int j = 0; j < 8; ++j)
        qf.u[j] = *(const unsigned int*)(qrow + kmapA(j, half));

    // ---- scores: S[r][j] = q_r . k_j  (q already scaled) ----
    const __bf16* kbase = kb + (size_t)i * L_DIM * C_DIM + h * D_DIM + kmapB(0, half);
    for (int jt = 0; jt < 20; ++jt) {
        const __bf16* krow = kbase + (size_t)(jt * 16 + lane15) * C_DIM;
        if (jt + 1 < 20)
            __builtin_prefetch(krow + 16 * C_DIM, 0, 1);   // global_prefetch_b8
        FragB kf;
#pragma unroll
        for (int j = 0; j < 8; ++j)
            kf.u[j] = *(const unsigned int*)(krow + 2 * j);
        v8f acc = {};
        acc = wmma_bf16(qf, kf, acc);
#pragma unroll
        for (int vv = 0; vv < 8; ++vv)
            S[wave][vv + 8 * half][jt * 16 + lane15] = acc[vv];
    }

    // ---- softmax over j (lane pair {l, l+16} splits the 320 columns) ----
    {
        const int r = lane15;
        float m = -3.0e38f;
        for (int c = half; c < L_DIM; c += 2)
            m = fmaxf(m, S[wave][r][c]);
        m = fmaxf(m, __shfl_xor(m, 16, 32));
        float sum = 0.0f;
        for (int c = half; c < L_DIM; c += 2) {
            float e = __expf(S[wave][r][c] - m);
            sum += e;
            P[wave][r][c] = (__bf16)e;     // unnormalized; 1/sum folded later
        }
        sum += __shfl_xor(sum, 16, 32);
        if (half == 0) Inv[wave][r] = __builtin_amdgcn_rcpf(sum);
    }

    // ---- O = P @ V (kt-outer: one P fragment feeds both N-tiles) ----
    v8f acc0 = {};
    v8f acc1 = {};
    const __bf16* vrow0 =
        vT + (size_t)((i * H_DIM + h) * D_DIM + lane15) * L_DIM + kmapB(0, half);
    const __bf16* vrow1 = vrow0 + (size_t)16 * L_DIM;
#pragma unroll
    for (int kt = 0; kt < 10; ++kt) {
        FragB pf, vf0, vf1;
#pragma unroll
        for (int j = 0; j < 8; ++j)
            pf.u[j] = *(const unsigned int*)&P[wave][lane15][kt * 32 + kmapA(j, half)];
#pragma unroll
        for (int j = 0; j < 8; ++j)
            vf0.u[j] = *(const unsigned int*)(vrow0 + kt * 32 + 2 * j);
#pragma unroll
        for (int j = 0; j < 8; ++j)
            vf1.u[j] = *(const unsigned int*)(vrow1 + kt * 32 + 2 * j);
        acc0 = wmma_bf16(pf, vf0, acc0);
        acc1 = wmma_bf16(pf, vf1, acc1);
    }
    __bf16* orow = ob + (size_t)(i * L_DIM + qbase) * C_DIM + h * D_DIM + lane15;
#pragma unroll
    for (int vv = 0; vv < 8; ++vv) {
        const int M = vv + 8 * half;
        const float inv = Inv[wave][M];
        orow[(size_t)M * C_DIM]      = (__bf16)(acc0[vv] * inv);
        orow[(size_t)M * C_DIM + 16] = (__bf16)(acc1[vv] * inv);
    }
}

// ---------------------------------------------------------------------------
// Kernel 3: out = (attn_out @ Wo + bo) * gate, fp32 store.
// One block = one 16-row tile; 8 waves, wave w = N-tile w; 4 K-chunk WMMA.
// ---------------------------------------------------------------------------
__global__ void __launch_bounds__(256) k_out(
    const __bf16* __restrict__ ob, const __bf16* __restrict__ WpO,
    const float* __restrict__ bo, const float* __restrict__ gate,
    float* __restrict__ out) {

    __shared__ __bf16 Os[16 * 128];

    const int rowbase = blockIdx.x * 16;
    {   // one uint4 per thread copies the whole bf16 tile
        const uint4* src = (const uint4*)(ob + (size_t)rowbase * 128);
        ((uint4*)Os)[threadIdx.x] = src[threadIdx.x];
    }
    __syncthreads();

    const int wave = threadIdx.x >> 5;
    const int lane = threadIdx.x & 31;
    const int lane15 = lane & 15;
    const int half = lane >> 4;
    const int c = wave * 16 + lane15;        // output channel

    const float b = bo[c];
    v8f acc = {};
#pragma unroll
    for (int z = 0; z < 8; ++z) acc[z] = b;

#pragma unroll
    for (int kt = 0; kt < 4; ++kt) {
        FragB af, bf;
#pragma unroll
        for (int j = 0; j < 8; ++j)
            af.u[j] = *(const unsigned int*)&Os[lane15 * 128 + kt * 32 + kmapA(j, half)];
#pragma unroll
        for (int j = 0; j < 8; ++j)
            bf.u[j] = *(const unsigned int*)&WpO[(size_t)c * 128 + kt * 32 + kmapB(j, half)];
        acc = wmma_bf16(af, bf, acc);
    }

#pragma unroll
    for (int vv = 0; vv < 8; ++vv) {
        const size_t idx = (size_t)(rowbase + vv + 8 * half) * 128 + c;
        out[idx] = acc[vv] * gate[idx];
    }
}

// ---------------------------------------------------------------------------
// Workspace layout (bytes)
// ---------------------------------------------------------------------------
static const size_t WP_OFF   = 0;                                // 5*128*128 bf16
static const size_t QB_OFF   = WP_OFF + 5 * 128 * 128 * 2;       // 26,214,400 B each
static const size_t KB_OFF   = QB_OFF + (size_t)NROWS * C_DIM * 2;
static const size_t VT_OFF   = KB_OFF + (size_t)NROWS * C_DIM * 2;
static const size_t OB_OFF   = VT_OFF + (size_t)NROWS * C_DIM * 2;
static const size_t GATE_OFF = OB_OFF + (size_t)NROWS * C_DIM * 2;

extern "C" void kernel_launch(void* const* d_in, const int* in_sizes, int n_in,
                              void* d_out, int out_size, void* d_ws, size_t ws_size,
                              hipStream_t stream) {
    (void)in_sizes; (void)n_in; (void)out_size; (void)ws_size;

    const float* pair = (const float*)d_in[0];
    const float* Wq = (const float*)d_in[1];  const float* bq = (const float*)d_in[2];
    const float* Wk = (const float*)d_in[3];  const float* bk = (const float*)d_in[4];
    const float* Wv = (const float*)d_in[5];  const float* bv = (const float*)d_in[6];
    const float* Wo = (const float*)d_in[7];  const float* bo = (const float*)d_in[8];
    const float* Wg = (const float*)d_in[9];  const float* bg = (const float*)d_in[10];

    char* ws = (char*)d_ws;
    __bf16* Wp   = (__bf16*)(ws + WP_OFF);
    __bf16* qb   = (__bf16*)(ws + QB_OFF);
    __bf16* kb   = (__bf16*)(ws + KB_OFF);
    __bf16* vT   = (__bf16*)(ws + VT_OFF);
    __bf16* ob   = (__bf16*)(ws + OB_OFF);
    float*  gate = (float*)(ws + GATE_OFF);

    k_weights<<<dim3(128, 5), 128, 0, stream>>>(Wq, Wk, Wv, Wg, Wo, Wp);
    k_proj<<<NTILES, 256, 0, stream>>>(pair, Wp, bq, bk, bv, bg, qb, kb, vT, gate);
    k_attn<<<dim3(10, H_DIM, L_DIM), 64, 0, stream>>>(qb, kb, vT, ob);
    k_out<<<NTILES, 256, 0, stream>>>(ob, Wp + 4 * 16384, bo, gate, (float*)d_out);
}
